// SelfAttention_867583394282
// MI455X (gfx1250) — compile-verified, matrix operand order checked by reading, not run on hardware
//
#include <hip/hip_runtime.h>

#define N_B   2
#define L_SEQ 2048
#define EMBED 1024
#define HEADS 16
#define HDIM  64

typedef __attribute__((ext_vector_type(16))) _Float16 v16h;
typedef __attribute__((ext_vector_type(8)))  _Float16 v8h;
typedef __attribute__((ext_vector_type(4)))  _Float16 v4h;
typedef __attribute__((ext_vector_type(8)))  float    v8f;
typedef __attribute__((ext_vector_type(4)))  float    v4f;

#if __has_builtin(__builtin_amdgcn_exp2f)
#define EXP2F(x) __builtin_amdgcn_exp2f(x)
#else
#define EXP2F(x) exp2f(x)
#endif

static __device__ __forceinline__ v8f wmma_f16(v16h a, v16h b, v8f c) {
  // D = A(16x32 f16) x B(32x16 f16) + C(16x16 f32)
  return __builtin_amdgcn_wmma_f32_16x16x32_f16(false, a, false, b, (short)0, c,
                                                false, false);
}

// Load a 16-half A/B operand slice for one lane.
// Layout: halves 0..7 = row[kc+hi8 .. +7], halves 8..15 = row[kc+hi8+16 .. +23]
// (hi8 = 8 for lanes 16..31), matching the CDNA5 16-bit A/B VGPR layout.
static __device__ __forceinline__ v16h ldab_h(const _Float16* __restrict__ row,
                                              int kc, int hi8) {
  v8h x0 = *(const v8h*)(row + kc + hi8);
  v8h x1 = *(const v8h*)(row + kc + hi8 + 16);
  v16h r;
#pragma unroll
  for (int i = 0; i < 8; ++i) { r[i] = x0[i]; r[i + 8] = x1[i]; }
  return r;
}

static __device__ __forceinline__ v16h ldab_f(const float* __restrict__ row,
                                              int kc, int hi8) {
  v8f x0 = *(const v8f*)(row + kc + hi8);
  v8f x1 = *(const v8f*)(row + kc + hi8 + 16);
  v16h r;
#pragma unroll
  for (int i = 0; i < 8; ++i) { r[i] = (_Float16)x0[i]; r[i + 8] = (_Float16)x1[i]; }
  return r;
}

// ---------------------------------------------------------------- Wo -> f16
__global__ void __launch_bounds__(256) cvt_wo(const float* __restrict__ Wo,
                                              _Float16* __restrict__ Wo16) {
  const int i = blockIdx.x * blockDim.x + threadIdx.x;  // over (1024*1024)/4
  v4f x = ((const v4f*)Wo)[i];
  v4h y;
#pragma unroll
  for (int r = 0; r < 4; ++r) y[r] = (_Float16)x[r];
  ((v4h*)Wo16)[i] = y;
}

// ------------------------------------------------------- QKV projections
// mode 0: Q -> Qp[nh][l][d] (scaled by log2e/8), transposed product (A=Wq,B=Xt)
// mode 1: K -> Kp[nh][l][d], transposed product
// mode 2: V -> Vt[nh][d][l], normal product (A=X, B=Wv^T)
__global__ void __launch_bounds__(128) proj_qkv(
    const float* __restrict__ vin, const float* __restrict__ kin,
    const float* __restrict__ qin, const float* __restrict__ Wv,
    const float* __restrict__ Wk, const float* __restrict__ Wq,
    _Float16* __restrict__ Qp, _Float16* __restrict__ Kp,
    _Float16* __restrict__ Vt) {
  const int mode = blockIdx.z;
  const float* in = (mode == 0) ? qin : (mode == 1) ? kin : vin;
  const float* W  = (mode == 0) ? Wq  : (mode == 1) ? Wk  : Wv;
  const int lane = threadIdx.x & 31;
  const int col  = lane & 15;
  const int hi8  = (lane & 16) ? 8 : 0;
  const int job  = blockIdx.x * (blockDim.x >> 5) + (threadIdx.x >> 5); // 0..1023
  const int nh  = job >> 5;   // 0..31 (n*16+h)
  const int ltg = job & 31;   // group of 4 query tiles
  const int n = nh >> 4, h = nh & 15;

  // Weight tiles wt[esub][kchunk]: A[m=e][k=d] = W[e][d]
  v16h wt[4][2];
#pragma unroll
  for (int e = 0; e < 4; ++e)
#pragma unroll
    for (int c = 0; c < 2; ++c)
      wt[e][c] = ldab_f(W + (size_t)(e * 16 + col) * HDIM, c * 32, hi8);

  const float qscale = 0.125f * 1.44269504088896340736f;  // 1/sqrt(64) * log2(e)

  for (int lt = 0; lt < 4; ++lt) {
    const int l0 = (ltg * 4 + lt) * 16;
    const float* xrow = in + ((size_t)n * L_SEQ + (l0 + col)) * EMBED + h * HDIM;
    v16h xt0 = ldab_f(xrow, 0, hi8);
    v16h xt1 = ldab_f(xrow, 32, hi8);
    if (mode < 2) {
      _Float16* outp = (mode == 0 ? Qp : Kp) +
                       ((size_t)nh * L_SEQ + (l0 + col)) * HDIM;
      const float s = (mode == 0) ? qscale : 1.0f;
#pragma unroll
      for (int e = 0; e < 4; ++e) {
        v8f c = {};
        c = wmma_f16(wt[e][0], xt0, c);   // C[e][l] = W X^T
        c = wmma_f16(wt[e][1], xt1, c);
        v8h hv;
#pragma unroll
        for (int r = 0; r < 8; ++r) hv[r] = (_Float16)(c[r] * s);
        *(v8h*)(outp + e * 16 + hi8) = hv;  // contiguous along e
      }
    } else {
#pragma unroll
      for (int e = 0; e < 4; ++e) {
        v8f c = {};
        c = wmma_f16(xt0, wt[e][0], c);   // C[l][e] = X W^T
        c = wmma_f16(xt1, wt[e][1], c);
        v8h hv;
#pragma unroll
        for (int r = 0; r < 8; ++r) hv[r] = (_Float16)c[r];
        // V^T store: [nh][d=e][l], contiguous along l
        *(v8h*)(Vt + ((size_t)nh * HDIM + e * 16 + col) * L_SEQ + l0 + hi8) = hv;
      }
    }
  }
}

// ------------------------------------------------------- flash attention
// One wave per (n,h,16-query tile). S^T = K Q^T; P^T feeds O^T = V^T P^T
// lane-locally (C-layout of S^T == B-layout of P^T). Q pre-scaled to log2 domain.
__global__ void __launch_bounds__(128) flash_attn(
    const _Float16* __restrict__ Qp, const _Float16* __restrict__ Kp,
    const _Float16* __restrict__ Vt, _Float16* __restrict__ AO) {
  const int lane = threadIdx.x & 31;
  const int col  = lane & 15;
  const int hi8  = (lane & 16) ? 8 : 0;
  const int job  = blockIdx.x * (blockDim.x >> 5) + (threadIdx.x >> 5); // 0..4095
  const int qtile = job & 127;
  const int nh    = job >> 7;
  const int n = nh >> 4, h = nh & 15;

  const _Float16* qrow = Qp + ((size_t)nh * L_SEQ + qtile * 16 + col) * HDIM;
  const v16h qb0 = ldab_h(qrow, 0, hi8);   // B operand: n=query col, k=dims
  const v16h qb1 = ldab_h(qrow, 32, hi8);

  const _Float16* kbase = Kp + (size_t)nh * L_SEQ * HDIM;
  const _Float16* vbase = Vt + (size_t)nh * HDIM * L_SEQ;

  v8f o0 = {}, o1 = {}, o2 = {}, o3 = {};   // O^T accumulators, 64 dims x 16 q
  float m_i = -3.0e38f, l_i = 0.f;

  for (int kb = 0; kb < L_SEQ; kb += 32) {
    const _Float16* kr0 = kbase + (size_t)(kb + col) * HDIM;
    const _Float16* kr1 = kbase + (size_t)(kb + 16 + col) * HDIM;
    v8f s0 = {}, s1 = {};                    // S^T tiles: keys x queries
    s0 = wmma_f16(ldab_h(kr0, 0, hi8), qb0, s0);
    s0 = wmma_f16(ldab_h(kr0, 32, hi8), qb1, s0);
    s1 = wmma_f16(ldab_h(kr1, 0, hi8), qb0, s1);
    s1 = wmma_f16(ldab_h(kr1, 32, hi8), qb1, s1);

    // online softmax per query column (lane-local + one xor-16 shuffle)
    float mx = m_i;
#pragma unroll
    for (int r = 0; r < 8; ++r) { mx = fmaxf(mx, s0[r]); mx = fmaxf(mx, s1[r]); }
    mx = fmaxf(mx, __shfl_xor(mx, 16, 32));
    const float alpha = EXP2F(m_i - mx);
    m_i = mx;

    float ls = 0.f;
    v16h pb;  // P^T in B-operand layout, built lane-locally from s0/s1
#pragma unroll
    for (int r = 0; r < 8; ++r) {
      float p0 = EXP2F(s0[r] - mx);
      float p1 = EXP2F(s1[r] - mx);
      ls += p0 + p1;
      pb[r]     = (_Float16)p0;
      pb[r + 8] = (_Float16)p1;
    }
    ls += __shfl_xor(ls, 16, 32);
    l_i = l_i * alpha + ls;

#pragma unroll
    for (int r = 0; r < 8; ++r) {
      o0[r] *= alpha; o1[r] *= alpha; o2[r] *= alpha; o3[r] *= alpha;
    }

    // O^T += V^T P^T ; A operand rows are dims of V^T (contiguous keys)
    o0 = wmma_f16(ldab_h(vbase + (size_t)( 0 + col) * L_SEQ + kb, 0, hi8), pb, o0);
    o1 = wmma_f16(ldab_h(vbase + (size_t)(16 + col) * L_SEQ + kb, 0, hi8), pb, o1);
    o2 = wmma_f16(ldab_h(vbase + (size_t)(32 + col) * L_SEQ + kb, 0, hi8), pb, o2);
    o3 = wmma_f16(ldab_h(vbase + (size_t)(48 + col) * L_SEQ + kb, 0, hi8), pb, o3);
  }

  const float inv = 1.0f / l_i;
  _Float16* arow = AO + ((size_t)n * L_SEQ + qtile * 16 + col) * EMBED + h * HDIM;
  v8f oo[4] = {o0, o1, o2, o3};
#pragma unroll
  for (int t = 0; t < 4; ++t) {
    v8h hv;
#pragma unroll
    for (int r = 0; r < 8; ++r) hv[r] = (_Float16)(oo[t][r] * inv);
    *(v8h*)(arow + t * 16 + hi8) = hv;   // contiguous along d
  }
}

// ------------------------------------------------------- output projection
// out[q][e] = sum_c AO[q][c] * Wo[e][c] + bo[e], transposed product so the
// f32 C tile stores as contiguous 32B per lane.
__global__ void __launch_bounds__(128) out_proj(
    const _Float16* __restrict__ AO, const _Float16* __restrict__ Wo16,
    const float* __restrict__ bo, float* __restrict__ out) {
  const int lane = threadIdx.x & 31;
  const int col  = lane & 15;
  const int hi8  = (lane & 16) ? 8 : 0;
  const int job  = blockIdx.x * (blockDim.x >> 5) + (threadIdx.x >> 5); // 0..4095
  const int eg    = job & 15;   // 4 e-tiles per wave
  const int qtile = job >> 4;   // 0..255 over N*L rows

  const _Float16* brow = AO + ((size_t)qtile * 16 + col) * EMBED;
  const _Float16* ar0 = Wo16 + (size_t)((eg * 4 + 0) * 16 + col) * EMBED;
  const _Float16* ar1 = Wo16 + (size_t)((eg * 4 + 1) * 16 + col) * EMBED;
  const _Float16* ar2 = Wo16 + (size_t)((eg * 4 + 2) * 16 + col) * EMBED;
  const _Float16* ar3 = Wo16 + (size_t)((eg * 4 + 3) * 16 + col) * EMBED;

  v8f c0 = {}, c1 = {}, c2 = {}, c3 = {};
#pragma unroll 4
  for (int kc = 0; kc < EMBED; kc += 32) {
    v16h b = ldab_h(brow, kc, hi8);          // AO^T operand, reused 4x
    c0 = wmma_f16(ldab_h(ar0, kc, hi8), b, c0);
    c1 = wmma_f16(ldab_h(ar1, kc, hi8), b, c1);
    c2 = wmma_f16(ldab_h(ar2, kc, hi8), b, c2);
    c3 = wmma_f16(ldab_h(ar3, kc, hi8), b, c3);
  }

  const size_t qg = (size_t)qtile * 16 + col;
  v8f cc[4] = {c0, c1, c2, c3};
#pragma unroll
  for (int t = 0; t < 4; ++t) {
    const int e0 = (eg * 4 + t) * 16 + hi8;
    v8f bv = *(const v8f*)(bo + e0);
#pragma unroll
    for (int r = 0; r < 8; ++r) cc[t][r] += bv[r];
    *(v8f*)(out + qg * EMBED + e0) = cc[t];   // contiguous along e
  }
}

// ---------------------------------------------------------------- launcher
extern "C" void kernel_launch(void* const* d_in, const int* in_sizes, int n_in,
                              void* d_out, int out_size, void* d_ws,
                              size_t ws_size, hipStream_t stream) {
  const float* vin = (const float*)d_in[0];
  const float* kin = (const float*)d_in[1];
  const float* qin = (const float*)d_in[2];
  const float* Wv  = (const float*)d_in[3];
  const float* Wk  = (const float*)d_in[4];
  const float* Wq  = (const float*)d_in[5];
  const float* Wo  = (const float*)d_in[6];
  const float* bo  = (const float*)d_in[7];
  float* out = (float*)d_out;

  char* ws = (char*)d_ws;
  const size_t SZ = (size_t)N_B * HEADS * L_SEQ * HDIM * sizeof(_Float16); // 8 MB
  _Float16* Qp   = (_Float16*)(ws);            // [nh][l][d]  (log2-scaled)
  _Float16* Kp   = (_Float16*)(ws + SZ);       // [nh][l][d]
  _Float16* Vt   = (_Float16*)(ws + 2 * SZ);   // [nh][d][l]
  _Float16* AO   = (_Float16*)(ws + 3 * SZ);   // [n][l][embed]
  _Float16* Wo16 = (_Float16*)(ws + 4 * SZ);   // [e][c] f16

  cvt_wo<<<1024, 256, 0, stream>>>(Wo, Wo16);

  dim3 gproj(256, 1, 3);  // z selects Q/K/V
  proj_qkv<<<gproj, 128, 0, stream>>>(vin, kin, qin, Wv, Wk, Wq, Qp, Kp, Vt);

  flash_attn<<<1024, 128, 0, stream>>>(Qp, Kp, Vt, AO);

  out_proj<<<1024, 128, 0, stream>>>(AO, Wo16, bo, out);
}
